// Attention_23940147708094
// MI455X (gfx1250) — matrix operand order repaired and checked by reading, not compiled, not run on hardware
//
#include <hip/hip_runtime.h>
#include <hip/hip_bf16.h>

// ---------------------------------------------------------------------------
// RoPE multi-head attention for MI455X (gfx1250), bf16 WMMA pipeline.
// B=2, L=2048, D=1024, 16 heads, hd=64.  Compute-bound -> everything through
// v_wmma_f32_16x16x32_bf16 with f32 accumulation.  All LDS fragment gathers
// are 16B-contiguous (ds_load_b128 pairs); tile staging uses the gfx1250
// async global->LDS DMA path (ASYNCcnt).  proj_gemm uses 32x64 wave tiles
// (8 wmma per 12 ds_load_b128) to keep LDS issue off the critical path.
// ---------------------------------------------------------------------------

typedef __bf16 bf16_t;
typedef __attribute__((ext_vector_type(16))) __bf16 v16bf;
typedef __attribute__((ext_vector_type(8)))  float  v8f;
typedef __attribute__((ext_vector_type(4)))  int    v4i;

union FragB16 { v16bf v; bf16_t e[16]; uint4 q[2]; };
union FragF8  { v8f  v; float  e[8];  };

__device__ inline v8f wmma_bf16(v16bf a, v16bf b, v8f c) {
  // (neg_a, A, neg_b, B, c_mod, C, reuse_a, reuse_b)
  return __builtin_amdgcn_wmma_f32_16x16x32_bf16(false, a, false, b, (short)0, c,
                                                 false, false);
}

// 16-byte global -> LDS copy via gfx1250 async-to-LDS DMA when available.
// Probed signature: (v4i AS1* src, v4i AS3* dst, imm offset, imm cpol).
#if defined(__has_builtin)
#if __has_builtin(__builtin_amdgcn_global_load_async_to_lds_b128)
#define ASYNC_COPY 1
#endif
#endif

__device__ inline void cp16(const bf16_t* __restrict__ g, bf16_t* __restrict__ l) {
#ifdef ASYNC_COPY
  __builtin_amdgcn_global_load_async_to_lds_b128(
      (__attribute__((address_space(1))) v4i*)g,
      (__attribute__((address_space(3))) v4i*)l, 0, 0);
#else
  *(uint4*)l = *(const uint4*)g;
#endif
}

__device__ inline void tile_wait() {
#ifdef ASYNC_COPY
  asm volatile("s_wait_asynccnt 0x0" ::: "memory");
#endif
}

constexpr int Dm     = 1024;
constexpr int Lm     = 2048;
constexpr int HD     = 64;
constexpr int NHEADS = 16;
constexpr int Bm     = 2;
constexpr int MrowsC = Bm * Lm;   // 4096

// ---------------------------------------------------------------------------
// f32 -> bf16 conversion
// ---------------------------------------------------------------------------
__global__ void cvt_bf16(const float* __restrict__ s, bf16_t* __restrict__ d, int n) {
  int i = blockIdx.x * blockDim.x + threadIdx.x;
  if (i < n) d[i] = (bf16_t)s[i];
}

// ---------------------------------------------------------------------------
// out = A(Mx1024, bf16) @ W^T (W is [N,K] row-major, bf16)
// mode 0: Q proj + RoPE -> headed bf16 [b][h][l][hd]
// mode 1: K proj + RoPE -> headed bf16 [b][h][l][hd]
// mode 2: V proj        -> TRANSPOSED headed bf16 [b][h][hd][l]
// mode 3: O proj        -> f32 [b*l][1024] (final output)
// Block: 128 threads (4 waves). Block tile 128x64, K-step 32.
// Each wave computes a 32x64 tile: 2 A-fragments x 4 B-fragments = 8 wmma.
// B tile is stored [n][k] (== W row-major) so B-fragments read contiguously.
// ---------------------------------------------------------------------------
__global__ __launch_bounds__(128) void proj_gemm(
    const bf16_t* __restrict__ A, const bf16_t* __restrict__ W,
    const float* __restrict__ freqs, bf16_t* __restrict__ outH,
    float* __restrict__ outF, int mode) {
  __shared__ bf16_t As[128][40];  // [m][k], pitch 80B (16B multiple)
  __shared__ bf16_t Bs[64][40];   // [n][k], pitch 80B

  const int t    = threadIdx.x;
  const int wave = t >> 5, lane = t & 31, l = lane & 15, hi = lane >> 4;
  const int rowBase = blockIdx.x * 128, colBase = blockIdx.y * 64;

  const v8f z8 = {0.f, 0.f, 0.f, 0.f, 0.f, 0.f, 0.f, 0.f};
  v8f acc[2][4] = {{z8, z8, z8, z8}, {z8, z8, z8, z8}};

  const int blr = t >> 1, blc0 = (t & 1) * 16;   // B tile loader coords

  for (int kb = 0; kb < Dm; kb += 32) {
    // A tile: 128 rows x 32 k, one row per thread (4 x 16B)
    const bf16_t* asrc = A + (size_t)(rowBase + t) * Dm + kb;
    // B tile: 64 rows x 32 k, half row per thread (2 x 16B)
    const bf16_t* bsrc = W + (size_t)(colBase + blr) * Dm + kb + blc0;
    __builtin_prefetch(asrc + 32, 0, 3);
    __builtin_prefetch(bsrc + 32, 0, 3);
    cp16(asrc,      &As[t][0]);
    cp16(asrc + 8,  &As[t][8]);
    cp16(asrc + 16, &As[t][16]);
    cp16(asrc + 24, &As[t][24]);
    cp16(bsrc,      &Bs[blr][blc0]);
    cp16(bsrc + 8,  &Bs[blr][blc0 + 8]);
    tile_wait();
    __syncthreads();

    // Two A fragments 16x32: rows wave*32 + {0,16} + l
    FragB16 af[2];
#pragma unroll
    for (int g = 0; g < 2; ++g) {
      const bf16_t* arow = &As[wave * 32 + g * 16 + l][0];
      af[g].q[0] = *(const uint4*)(arow + hi * 16);
      af[g].q[1] = *(const uint4*)(arow + hi * 16 + 8);
    }

#pragma unroll
    for (int j = 0; j < 4; ++j) {
      // B fragment 32x16: lane (l,hi) reads col l, k = hi*16..+15 (contiguous)
      FragB16 bfx;
      const bf16_t* brow = &Bs[j * 16 + l][0];
      bfx.q[0] = *(const uint4*)(brow + hi * 16);
      bfx.q[1] = *(const uint4*)(brow + hi * 16 + 8);
      acc[0][j] = wmma_bf16(af[0].v, bfx.v, acc[0][j]);
      acc[1][j] = wmma_bf16(af[1].v, bfx.v, acc[1][j]);
    }
    __syncthreads();
  }

  // Epilogue: element r of acc[g][j] is (row = 32w + 16g + r + 8hi, col = 16j + l)
#pragma unroll
  for (int g = 0; g < 2; ++g) {
#pragma unroll
    for (int j = 0; j < 4; ++j) {
      FragF8 c; c.v = acc[g][j];
#pragma unroll
      for (int r = 0; r < 8; ++r) {
        const int row = rowBase + wave * 32 + g * 16 + r + 8 * hi;
        const int col = colBase + j * 16 + l;
        float val = c.e[r];
        if (mode == 3) {
          outF[(size_t)row * Dm + col] = val;
        } else {
          const int bb = row >> 11, pos = row & (Lm - 1);
          const int h  = col >> 6,  hd  = col & (HD - 1);
          if (mode <= 1) {
            // RoPE: rotation partner (col^1) is in the adjacent lane, same row
            const float partner = __shfl_xor(val, 1, 32);
            const float* f = freqs + ((size_t)pos * (HD / 2) + (hd >> 1)) * 2;
            const float cs = f[0], sn = f[1];
            val = (hd & 1) ? (partner * sn + val * cs)   // imag part
                           : (val * cs - partner * sn);  // real part
            outH[(((size_t)bb * NHEADS + h) * Lm + pos) * HD + hd] = (bf16_t)val;
          } else {
            // V: store transposed [b][h][hd][pos] so flash PV B-fragments are
            // key-contiguous in LDS.
            outH[(((size_t)bb * NHEADS + h) * HD + hd) * Lm + pos] = (bf16_t)val;
          }
        }
      }
    }
  }
}

// ---------------------------------------------------------------------------
// Flash attention: one wave per 16-row Q tile, 8 waves / block (128 q rows of
// one (b,head)).  32-key blocks: QK^T (4 wmma) -> online softmax -> PV (4 wmma).
// grid = (L/128, B*NHEADS), block = 256.  vT is [b][h][hd][pos].
// ---------------------------------------------------------------------------
__global__ __launch_bounds__(256) void flash_attn(
    const bf16_t* __restrict__ q, const bf16_t* __restrict__ k,
    const bf16_t* __restrict__ vT, bf16_t* __restrict__ attnh) {
  __shared__ bf16_t Ks[32][72];      // [key][hd], pitch 144B
  __shared__ bf16_t Vt[64][40];      // [vd][key], pitch 80B
  __shared__ bf16_t Ps[8][16][40];   // per-wave P staging, pitch 80B

  const int t    = threadIdx.x;
  const int wave = t >> 5, lane = t & 31, l = lane & 15, hi = lane >> 4;
  const int bh   = blockIdx.y;                        // b*16 + h
  const int qBase = blockIdx.x * 128 + wave * 16;
  const size_t headOff = (size_t)bh * Lm * HD;

  // Q fragments: A-matrix 16x64 = two 16x32 fragments (hd 0..31, 32..63)
  FragB16 qa0, qa1;
  {
    const bf16_t* qrow = q + headOff + (size_t)(qBase + l) * HD;
    qa0.q[0] = *(const uint4*)(qrow + hi * 16);
    qa0.q[1] = *(const uint4*)(qrow + hi * 16 + 8);
    qa1.q[0] = *(const uint4*)(qrow + 32 + hi * 16);
    qa1.q[1] = *(const uint4*)(qrow + 40 + hi * 16);
  }

  float Mr[8], Sr[8];
#pragma unroll
  for (int r = 0; r < 8; ++r) { Mr[r] = -1e30f; Sr[r] = 0.f; }
  const v8f z8 = {0.f, 0.f, 0.f, 0.f, 0.f, 0.f, 0.f, 0.f};
  v8f o[4] = {z8, z8, z8, z8};

  const int kEnd = blockIdx.x * 128 + 128;      // block-uniform trip count
  const int kr   = t >> 3, kc = (t & 7) * 8;    // K-tile loader: 32x64
  const int vr   = t >> 2, vc = (t & 3) * 8;    // V-tile loader: 64x32

  for (int kb = 0; kb < kEnd; kb += 32) {
    cp16(k  + headOff + (size_t)(kb + kr) * HD + kc, &Ks[kr][kc]);
    cp16(vT + headOff + (size_t)vr * Lm + kb + vc,   &Vt[vr][vc]);
    tile_wait();
    __syncthreads();

    if (kb < qBase + 16) {   // wave-uniform causal skip; barriers stay uniform
      // S = Q K^T  (two 16-key subtiles, each accumulating over hd=64)
      FragF8 s[2];
#pragma unroll
      for (int ts = 0; ts < 2; ++ts) {
        FragB16 b0, b1;   // B[kdim=hd][n=key]: key = ts*16 + l, hd contiguous
        const bf16_t* krow = &Ks[ts * 16 + l][0];
        b0.q[0] = *(const uint4*)(krow + hi * 16);
        b0.q[1] = *(const uint4*)(krow + hi * 16 + 8);
        b1.q[0] = *(const uint4*)(krow + 32 + hi * 16);
        b1.q[1] = *(const uint4*)(krow + 40 + hi * 16);
        v8f sv = z8;
        sv = wmma_bf16(qa0.v, b0.v, sv);
        sv = wmma_bf16(qa1.v, b1.v, sv);
        s[ts].v = sv;
      }
      // scale 1/sqrt(64) + causal mask
#pragma unroll
      for (int ts = 0; ts < 2; ++ts) {
#pragma unroll
        for (int r = 0; r < 8; ++r) {
          const int key = kb + ts * 16 + l;
          const int row = qBase + r + 8 * hi;
          const float val = s[ts].e[r] * 0.125f;
          s[ts].e[r] = (key > row) ? -1e30f : val;
        }
      }
      // online softmax: row stats live across the 16 lanes owning a C-row
      float nM[8], alpha[8];
#pragma unroll
      for (int r = 0; r < 8; ++r) {
        float m = fmaxf(s[0].e[r], s[1].e[r]);
        m = fmaxf(m, __shfl_xor(m, 1, 32));
        m = fmaxf(m, __shfl_xor(m, 2, 32));
        m = fmaxf(m, __shfl_xor(m, 4, 32));
        m = fmaxf(m, __shfl_xor(m, 8, 32));
        nM[r]    = fmaxf(Mr[r], m);
        alpha[r] = __expf(Mr[r] - nM[r]);
        Mr[r]    = nM[r];
      }
      // P = exp(s - newM): store to LDS (C-layout), accumulate row sums
#pragma unroll
      for (int r = 0; r < 8; ++r) {
        const float p0 = __expf(s[0].e[r] - nM[r]);
        const float p1 = __expf(s[1].e[r] - nM[r]);
        Ps[wave][r + 8 * hi][l]      = (bf16_t)p0;
        Ps[wave][r + 8 * hi][16 + l] = (bf16_t)p1;
        float rs = p0 + p1;
        rs += __shfl_xor(rs, 1, 32);
        rs += __shfl_xor(rs, 2, 32);
        rs += __shfl_xor(rs, 4, 32);
        rs += __shfl_xor(rs, 8, 32);
        Sr[r] = Sr[r] * alpha[r] + rs;
      }
      // rescale running output accumulator
#pragma unroll
      for (int j = 0; j < 4; ++j) {
        FragF8 oj; oj.v = o[j];
#pragma unroll
        for (int r = 0; r < 8; ++r) oj.e[r] *= alpha[r];
        o[j] = oj.v;
      }
      // intra-wave cross-lane LDS visibility: wait DS ops, no barrier needed
      asm volatile("s_wait_dscnt 0x0" ::: "memory");
      // P as A-matrix 16x32 fragment (contiguous per lane)
      FragB16 pa;
      const bf16_t* prow = &Ps[wave][l][0];
      pa.q[0] = *(const uint4*)(prow + hi * 16);
      pa.q[1] = *(const uint4*)(prow + hi * 16 + 8);
      // O += P V  (4 output column subtiles); V fragment key-contiguous
#pragma unroll
      for (int j = 0; j < 4; ++j) {
        FragB16 vb;   // B[kdim=key][n=vd]: key = hi*16 + i, vd = j*16 + l
        const bf16_t* vrow = &Vt[j * 16 + l][0];
        vb.q[0] = *(const uint4*)(vrow + hi * 16);
        vb.q[1] = *(const uint4*)(vrow + hi * 16 + 8);
        o[j] = wmma_bf16(pa.v, vb.v, o[j]);
      }
    }
    __syncthreads();
  }

  // normalize and write merged-head bf16 activations [b][pos][h*64+vd]
  const int bb = bh >> 4, h = bh & 15;
#pragma unroll
  for (int j = 0; j < 4; ++j) {
    FragF8 oj; oj.v = o[j];
#pragma unroll
    for (int r = 0; r < 8; ++r) {
      const int pos = qBase + r + 8 * hi;
      const int vd  = j * 16 + l;
      const float val = oj.e[r] / Sr[r];
      attnh[((size_t)bb * Lm + pos) * Dm + h * HD + vd] = (bf16_t)val;
    }
  }
}

// ---------------------------------------------------------------------------
// Host-side orchestration (graph-capture safe: only kernel launches on stream)
// ---------------------------------------------------------------------------
extern "C" void kernel_launch(void* const* d_in, const int* in_sizes, int n_in,
                              void* d_out, int out_size, void* d_ws, size_t ws_size,
                              hipStream_t stream) {
  (void)in_sizes; (void)n_in; (void)out_size; (void)ws_size;

  const float* x     = (const float*)d_in[0];
  const float* freqs = (const float*)d_in[1];
  // d_in[2] = attention_mask (unused: causal mask applied analytically)
  const float* Wq = (const float*)d_in[3];
  const float* Wk = (const float*)d_in[4];
  const float* Wv = (const float*)d_in[5];
  const float* Wo = (const float*)d_in[6];
  float* out = (float*)d_out;

  char*  ws  = (char*)d_ws;
  size_t off = 0;
  auto alloc = [&](size_t bytes) {
    char* p = ws + off;
    off += (bytes + 255) & ~(size_t)255;
    return p;
  };
  bf16_t* xh  = (bf16_t*)alloc((size_t)MrowsC * Dm * sizeof(bf16_t));
  bf16_t* wqh = (bf16_t*)alloc((size_t)Dm * Dm * sizeof(bf16_t));
  bf16_t* wkh = (bf16_t*)alloc((size_t)Dm * Dm * sizeof(bf16_t));
  bf16_t* wvh = (bf16_t*)alloc((size_t)Dm * Dm * sizeof(bf16_t));
  bf16_t* woh = (bf16_t*)alloc((size_t)Dm * Dm * sizeof(bf16_t));
  const size_t headedElems = (size_t)Bm * NHEADS * Lm * HD;
  bf16_t* qh  = (bf16_t*)alloc(headedElems * sizeof(bf16_t));
  bf16_t* kh  = (bf16_t*)alloc(headedElems * sizeof(bf16_t));
  bf16_t* vth = (bf16_t*)alloc(headedElems * sizeof(bf16_t));  // transposed V
  bf16_t* ah  = (bf16_t*)alloc((size_t)MrowsC * Dm * sizeof(bf16_t));

  const int nX = MrowsC * Dm;
  const int nW = Dm * Dm;
  cvt_bf16<<<(nX + 255) / 256, 256, 0, stream>>>(x,  xh,  nX);
  cvt_bf16<<<(nW + 255) / 256, 256, 0, stream>>>(Wq, wqh, nW);
  cvt_bf16<<<(nW + 255) / 256, 256, 0, stream>>>(Wk, wkh, nW);
  cvt_bf16<<<(nW + 255) / 256, 256, 0, stream>>>(Wv, wvh, nW);
  cvt_bf16<<<(nW + 255) / 256, 256, 0, stream>>>(Wo, woh, nW);

  const dim3 gGemm(MrowsC / 128, Dm / 64);   // 32 x 16 blocks
  proj_gemm<<<gGemm, 128, 0, stream>>>(xh, wqh, freqs, qh,  nullptr, 0);
  proj_gemm<<<gGemm, 128, 0, stream>>>(xh, wkh, freqs, kh,  nullptr, 1);
  proj_gemm<<<gGemm, 128, 0, stream>>>(xh, wvh, freqs, vth, nullptr, 2);

  flash_attn<<<dim3(Lm / 128, Bm * NHEADS), 256, 0, stream>>>(qh, kh, vth, ah);

  proj_gemm<<<gGemm, 128, 0, stream>>>(ah, woh, freqs, nullptr, out, 3);
}
